// MambaEncoder_9440338116670
// MI455X (gfx1250) — matrix operand order repaired and checked by reading, not compiled, not run on hardware
//
#include <hip/hip_runtime.h>

#define BATCH 2048
#define EPSq 1e-5f

typedef __attribute__((ext_vector_type(16))) _Float16 v16h;
typedef __attribute__((ext_vector_type(8)))  _Float16 v8h;
typedef __attribute__((ext_vector_type(8)))  float    v8f;

// Padded all-LDS GEMM via v_wmma_f32_16x16x32_f16 (wave32).
// A: (M x K) f16 row-major, K-contiguous.  Bt: (N x K) f16 row-major
// (B stored K-contiguous per output column).  C: (M x N), ld = N.
// M,N multiples of 16; K multiple of 32.  Branchless; fragment gathers are
// aligned 16B LDS vector loads (ds_load_b128).
template<int M, int N, int K, bool ACC, typename TC>
__device__ __forceinline__ void wmma_gemm(const _Float16* __restrict__ A,
                                          const _Float16* __restrict__ Bt,
                                          TC* __restrict__ C, int tid)
{
  constexpr int MT = M / 16, NT = N / 16;
  const int lane = tid & 31;
  const int wave = tid >> 5;
  const int hi   = lane >> 4;        // half-wave select
  const int mrow = lane & 15;
  for (int tile = wave; tile < MT * NT; tile += 18) {
    const int m0 = (tile / NT) * 16;
    const int n0 = (tile % NT) * 16;
    const _Float16* Ar = A  + (m0 + mrow) * K + hi * 8;
    const _Float16* Br = Bt + (n0 + mrow) * K + hi * 8;
    TC* Cc = C + n0 + mrow;
    v8f acc;
#pragma unroll
    for (int r = 0; r < 8; ++r)
      acc[r] = ACC ? (float)Cc[(m0 + r + hi * 8) * N] : 0.0f;
#pragma unroll
    for (int k0 = 0; k0 < K; k0 += 32) {
      v8h a0 = *(const v8h*)(Ar + k0);        // K = k0+hi*8 .. +7
      v8h a1 = *(const v8h*)(Ar + k0 + 16);   // K = k0+16+hi*8 .. +7
      v8h b0 = *(const v8h*)(Br + k0);        // K = k0+hi*8 .. +7
      v8h b1 = *(const v8h*)(Br + k0 + 16);   // K = k0+16+hi*8 .. +7
      v16h af, bf;
#pragma unroll
      for (int j = 0; j < 8; ++j) {
        af[j] = a0[j]; af[j + 8] = a1[j];
        bf[j] = b0[j]; bf[j + 8] = b1[j];
      }
      acc = __builtin_amdgcn_wmma_f32_16x16x32_f16(false, af, false, bf,
                                                   (short)0, acc, false, false);
    }
#pragma unroll
    for (int r = 0; r < 8; ++r)
      Cc[(m0 + r + hi * 8) * N] = (TC)acc[r];
  }
}

__device__ __forceinline__ float siluf(float v) { return v / (1.f + __expf(-v)); }

__global__ __launch_bounds__(576) void mamba_encoder_kernel(
    const float* __restrict__ x,
    const float* __restrict__ bn_gamma, const float* __restrict__ bn_beta,
    const float* __restrict__ bn_mean,  const float* __restrict__ bn_var,
    const float* __restrict__ ln_w,     const float* __restrict__ ln_b,
    const float* __restrict__ W_in,     const float* __restrict__ W_x,
    const float* __restrict__ W_dt,     const float* __restrict__ b_dt,
    const float* __restrict__ A_log,    const float* __restrict__ Dv,
    const float* __restrict__ W_out,
    const float* __restrict__ W1, const float* __restrict__ b1,
    const float* __restrict__ W2, const float* __restrict__ b2,
    const float* __restrict__ rv_w, const float* __restrict__ rv_b,
    const float* __restrict__ fw,
    float* __restrict__ out)
{
  // ---- LDS pool with manual overlay (62464 B total) ----
  __shared__ __align__(16) unsigned char smem[62464];
  float*    P_h    = (float*)(smem);              // 128x16 f32 (padded h)
  float*    P_mean = (float*)(smem + 8192);       // 9
  float*    P_std  = (float*)(smem + 8256);       // 9
  _Float16* P_z    = (_Float16*)(smem + 8320);    // 120x36 (z gate)
  unsigned char* S = smem + 16960;                // scratch region (45504 B)
  _Float16* P_xz   = (_Float16*)(S);              // 128x80
  _Float16* P_hnA  = (_Float16*)(S + 20480);      // 128x32
  _Float16* P_Win  = (_Float16*)(S + 28672);      // 80x32  (contig after hnA)
  _Float16* P_xiA  = (_Float16*)(S + 20480);      // 128x64 (aliases hnA+Win)
  _Float16* P_Wx   = (_Float16*)(S);              // 48x64  (aliases xz)
  _Float16* P_xdbl = (_Float16*)(S + 6144);       // 128x48
  _Float16* P_ysc  = (_Float16*)(S + 36864);      // 120x36 scan output
  _Float16* P_Wout = (_Float16*)(S);              // 16x64
  _Float16* P_hT   = (_Float16*)(S);              // 16x128 (head)
  _Float16* P_W1B  = (_Float16*)(S + 4096);       // 96x128
  _Float16* P_g1   = (_Float16*)(S + 28672);      // 16x96
  _Float16* P_W2B  = (_Float16*)(S);              // 32x96
  float*    P_g2   = (float*)(S + 6144);          // 16x32
  float*    P_lat  = (float*)(S + 8192);          // 216
  float*    P_Y    = (float*)(S + 9056);          // 24x5x2 DFT bins

  const int b   = blockIdx.x;
  const int tid = threadIdx.x;

  // ---- Phase 0: padded load of x row (clamped index + select, branchless) ----
  for (int i = tid; i < 2048; i += 576) {
    int m = i >> 4, c = i & 15;
    float v = x[b * 1080 + (m < 120 ? m : 119) * 9 + (c < 9 ? c : 8)];
    P_h[i] = (m < 120 && c < 9) ? v : 0.f;
  }
  __syncthreads();
  if (tid < 9) {
    float s = 0.f;
    for (int l = 0; l < 120; ++l) s += P_h[l * 16 + tid];
    float m = s * (1.f / 120.f);
    float v = 0.f;
    for (int l = 0; l < 120; ++l) { float d = P_h[l * 16 + tid] - m; v += d * d; }
    P_mean[tid] = m;
    P_std[tid]  = sqrtf(v * (1.f / 119.f)) + EPSq;   // reference std + EPS
  }
  __syncthreads();
  for (int i = tid; i < 1080; i += 576) {          // instance norm + batchnorm
    int l = i / 9, bar = i % 9, idx = l * 16 + bar;
    float xn = (P_h[idx] - P_mean[bar]) / P_std[bar];
    P_h[idx] = (xn - bn_mean[i]) * rsqrtf(bn_var[i] + EPSq) * bn_gamma[i] + bn_beta[i];
  }
  __syncthreads();

  // ---- Mamba layers ----
  for (int layer = 0; layer < 3; ++layer) {
    // (a) zero hnA + Win (contiguous 6656 halfs)
    for (int i = tid; i < 6656; i += 576) P_hnA[i] = (_Float16)0.f;
    __syncthreads();
    // (b) LayerNorm into hnA rows; stage W_in^T (Bt: 80x32)
    const float* lw = ln_w + layer * 9;
    const float* lb = ln_b + layer * 9;
    for (int l = tid; l < 120; l += 576) {
      float mu = 0.f;
      for (int j = 0; j < 9; ++j) mu += P_h[l * 16 + j];
      mu *= (1.f / 9.f);
      float var = 0.f;
      for (int j = 0; j < 9; ++j) { float d = P_h[l * 16 + j] - mu; var += d * d; }
      float rs = rsqrtf(var * (1.f / 9.f) + EPSq);
      for (int j = 0; j < 9; ++j)
        P_hnA[l * 32 + j] = (_Float16)((P_h[l * 16 + j] - mu) * rs * lw[j] + lb[j]);
    }
    for (int i = tid; i < 2560; i += 576) {        // Bt[n*32+k] = W_in[n][k]
      int n = i >> 5, k = i & 31;
      float v = W_in[layer * 648 + (n < 72 ? n : 71) * 9 + (k < 9 ? k : 8)];
      P_Win[i] = (_Float16)((n < 72 && k < 9) ? v : 0.f);
    }
    __syncthreads();
    // (c) xz = hn @ W_in^T : 128x80x32
    wmma_gemm<128, 80, 32, false>(P_hnA, P_Win, P_xz, tid);
    __syncthreads();
    // (d) zero xiA (over hnA/Win region)
    for (int i = tid; i < 8192; i += 576) P_xiA[i] = (_Float16)0.f;
    __syncthreads();
    // (e) split: xi = silu(xz[:, :36]) -> xiA; z -> P_z
    for (int i = tid; i < 4320; i += 576) {
      int l = i / 36, d = i % 36;
      float v = (float)P_xz[l * 80 + d];
      P_xiA[l * 64 + d] = (_Float16)siluf(v);
      P_z[i] = P_xz[l * 80 + 36 + d];
    }
    __syncthreads();
    // (f) stage W_x^T (Bt: 48x64) over dead xz region
    for (int i = tid; i < 3072; i += 576) {
      int n = i >> 6, k = i & 63;
      float v = W_x[layer * 1224 + (n < 34 ? n : 33) * 36 + (k < 36 ? k : 35)];
      P_Wx[i] = (_Float16)((n < 34 && k < 36) ? v : 0.f);
    }
    __syncthreads();
    // (g) x_dbl = xi @ W_x^T : 128x48x64
    wmma_gemm<128, 48, 64, false>(P_xiA, P_Wx, P_xdbl, tid);
    __syncthreads();
    // (h) selective scan.  thread=(d,n), 36x16=576.  The softplus(delta) is
    // per-(t,d): amortize it 16x by letting lane n of each 16-lane state
    // group compute delta for t=t0+n, then broadcasting via __shfl inside
    // the dependent loop.  Dependent chain per step: shfl, v_exp, 2 fma,
    // shfl_xor-tree reduce.
    {
      const int d = tid >> 4, n = tid & 15;
      const int grp = (tid & 31) & 16;            // lane group base in wave
      const float And = -__expf(A_log[layer * 576 + d * 16 + n]);
      const float w0  = W_dt[layer * 72 + d * 2 + 0];
      const float w1  = W_dt[layer * 72 + d * 2 + 1];
      const float bd  = b_dt[layer * 36 + d];
      float hst = 0.f;
      for (int t0 = 0; t0 < 120; t0 += 16) {
        int tt = t0 + n; tt = (tt < 119) ? tt : 119;       // clamp (lookahead)
        float dr0 = (float)P_xdbl[tt * 48 + 0];
        float dr1 = (float)P_xdbl[tt * 48 + 1];
        float arg = fmaf(dr0, w0, fmaf(dr1, w1, bd));
        float myD = (arg > 20.f) ? arg : log1pf(__expf(arg));  // softplus
        const int qmax = (120 - t0 < 16) ? (120 - t0) : 16;    // wave-uniform
        for (int q = 0; q < qmax; ++q) {
          const int t = t0 + q;
          float dlt = __shfl(myD, grp | q, 32);
          float u   = (float)P_xiA[t * 64 + d];
          float Bn  = (float)P_xdbl[t * 48 + 2 + n];
          float Cn  = (float)P_xdbl[t * 48 + 18 + n];
          hst = __expf(dlt * And) * hst + dlt * Bn * u;
          float p = hst * Cn;
          p += __shfl_xor(p, 1, 32);
          p += __shfl_xor(p, 2, 32);
          p += __shfl_xor(p, 4, 32);
          p += __shfl_xor(p, 8, 32);
          if (n == 0) P_ysc[t * 36 + d] = (_Float16)p;
        }
      }
    }
    __syncthreads();
    // (i) gating: xiA <- (y_scan + u*D) * silu(z); stage W_out^T (Bt 16x64)
    for (int i = tid; i < 4320; i += 576) {
      int l = i / 36, d = i % 36;
      float u  = (float)P_xiA[l * 64 + d];
      float z  = (float)P_z[i];
      float yv = (float)P_ysc[i] + u * Dv[layer * 36 + d];
      P_xiA[l * 64 + d] = (_Float16)(yv * siluf(z));
    }
    for (int i = tid; i < 1024; i += 576) {
      int n = i >> 6, k = i & 63;
      float v = W_out[layer * 324 + (n < 9 ? n : 8) * 36 + (k < 36 ? k : 35)];
      P_Wout[i] = (_Float16)((n < 9 && k < 36) ? v : 0.f);
    }
    __syncthreads();
    // (j) h += y @ W_out^T : 128x16x64, residual accumulate into f32 h
    wmma_gemm<128, 16, 64, true>(P_xiA, P_Wout, P_h, tid);
    __syncthreads();
  }

  // ---- Head ----
  // (k) stage h^T (16x128) and W1 (Bt 96x128)
  for (int i = tid; i < 2048; i += 576) {
    int m = i >> 7, k = i & 127;
    P_hT[i] = (_Float16)P_h[k * 16 + m];           // padding already zero
  }
  for (int i = tid; i < 12288; i += 576) {
    int n = i >> 7, k = i & 127;
    float v = W1[(n < 96 ? n : 95) * 120 + (k < 120 ? k : 119)];
    P_W1B[i] = (_Float16)((k < 120) ? v : 0.f);    // n < 96 always
  }
  __syncthreads();
  // (l) g1 = h^T @ W1^T : 16x96x128
  wmma_gemm<16, 96, 128, false>(P_hT, P_W1B, P_g1, tid);
  __syncthreads();
  // (m) g1 = leaky(g1 + b1)
  for (int i = tid; i < 1536; i += 576) {
    float v = (float)P_g1[i] + b1[i % 96];
    P_g1[i] = (_Float16)((v > 0.f) ? v : 0.01f * v);
  }
  __syncthreads();
  // (n) stage W2 (Bt 32x96)
  for (int i = tid; i < 3072; i += 576) {
    int n = i / 96, k = i % 96;
    float v = W2[(n < 24 ? n : 23) * 96 + k];
    P_W2B[i] = (_Float16)((n < 24) ? v : 0.f);
  }
  __syncthreads();
  // (o) g2 = g1 @ W2^T : 16x32x96
  wmma_gemm<16, 32, 96, false>(P_g1, P_W2B, P_g2, tid);
  __syncthreads();
  // (p) latent = (tanh(g2+b2)*rv_w + rv_b) * (std+EPS) + mean
  for (int j = tid; j < 216; j += 576) {
    int f = j / 9, c = j % 9;
    float hv = tanhf(P_g2[c * 32 + f] + b2[f]);    // swapaxes(1,2) of (9,24)
    int barIdx = j / 120;                          // mean/std layout bar*L+l
    float lat = (hv * rv_w[j] + rv_b[j]) * (P_std[barIdx] + EPSq) + P_mean[barIdx];
    P_lat[j] = lat;
    out[b * 216 + j] = lat;
  }
  __syncthreads();
  // (q) rfft(9) * w  -> 5 complex bins per NF row
  if (tid < 120) {
    int f = tid / 5, k = tid % 5;
    float xr = 0.f, xim = 0.f;
    for (int t = 0; t < 9; ++t) {
      float v   = P_lat[f * 9 + t];
      float ang = -6.283185307179586f * (float)(k * t) * (1.f / 9.f);
      xr  += v * __cosf(ang);
      xim += v * __sinf(ang);
    }
    float wr = 1.f / (1.f + __expf(-fw[f * 10 + k * 2 + 0]));
    float wi = fw[f * 10 + k * 2 + 1];
    P_Y[(f * 5 + k) * 2 + 0] = xr * wr - xim * wi;
    P_Y[(f * 5 + k) * 2 + 1] = xr * wi + xim * wr;
  }
  __syncthreads();
  // (r) irfft(9): DC imag ignored, 2*Re for k=1..4
  for (int j = tid; j < 216; j += 576) {
    int f = j / 9, t = j % 9;
    float acc = P_Y[(f * 5) * 2];
    for (int k = 1; k < 5; ++k) {
      float ang = 6.283185307179586f * (float)(k * t) * (1.f / 9.f);
      acc += 2.f * (P_Y[(f * 5 + k) * 2] * __cosf(ang)
                    - P_Y[(f * 5 + k) * 2 + 1] * __sinf(ang));
    }
    out[BATCH * 216 + b * 216 + j] = acc * (1.f / 9.f);
  }
}

extern "C" void kernel_launch(void* const* d_in, const int* in_sizes, int n_in,
                              void* d_out, int out_size, void* d_ws, size_t ws_size,
                              hipStream_t stream) {
  (void)in_sizes; (void)n_in; (void)out_size; (void)d_ws; (void)ws_size;
  mamba_encoder_kernel<<<BATCH, 576, 0, stream>>>(
      (const float*)d_in[0],  (const float*)d_in[1],  (const float*)d_in[2],
      (const float*)d_in[3],  (const float*)d_in[4],  (const float*)d_in[5],
      (const float*)d_in[6],  (const float*)d_in[7],  (const float*)d_in[8],
      (const float*)d_in[9],  (const float*)d_in[10], (const float*)d_in[11],
      (const float*)d_in[12], (const float*)d_in[13], (const float*)d_in[14],
      (const float*)d_in[15], (const float*)d_in[16], (const float*)d_in[17],
      (const float*)d_in[18], (const float*)d_in[19], (const float*)d_in[20],
      (float*)d_out);
}
